// SentBertSelfAttention_33698313404628
// MI455X (gfx1250) — matrix operand order repaired and checked by reading, not compiled, not run on hardware
//
#include <hip/hip_runtime.h>
#include <hip/hip_bf16.h>

// ---------------------------------------------------------------------------
// BERT self-attention (relative_key_query) for gfx1250 / MI455X.
// Pipeline:
//   1) cvt kernels: fp32 -> bf16 staging (X, W^T, dist_emb padded to 1024 rows)
//   2) qkv_proj:    WMMA bf16 GEMM -> Q,K [B,NH,S,HD] bf16, V^T [B,NH,HD,S] bf16
//   3) attn:        flash-attention per (b,h,16-query-tile) wave; positional
//                   bias via banded WMMA GEMMs (H, G) staged in LDS; next-tile
//                   K/V prefetch via global_prefetch_b8.
// Workspace: ~54 MB.
// ---------------------------------------------------------------------------

typedef __bf16 bf16_t;
typedef __attribute__((ext_vector_type(16))) __bf16 v16bf;
typedef __attribute__((ext_vector_type(8)))  __bf16 v8bf;
typedef __attribute__((ext_vector_type(8)))  float  v8f;

#define WMMA_BF16(a, b, c) \
  __builtin_amdgcn_wmma_f32_16x16x32_bf16(false, (a), false, (b), (short)0, (c), false, false)

#define LDS_SYNC() asm volatile("s_wait_dscnt 0" ::: "memory")

// A-fragment (16x32 bf16, row-major source): lane holds row (lane%16),
// K elements [2*c0..2*c0+15], c0 = 8*(lane/16).
static __device__ __forceinline__ v16bf load_a16(const bf16_t* __restrict__ p, int c0) {
  v8bf lo = *(const v8bf*)(p + 2 * c0);
  v8bf hi = *(const v8bf*)(p + 2 * c0 + 8);
  v16bf a;
#pragma unroll
  for (int i = 0; i < 8; ++i) { a[i] = lo[i]; a[i + 8] = hi[i]; }
  return a;
}

// ------------------------- conversion kernels ------------------------------

__global__ void cvt_bf16_kernel(const float* __restrict__ src,
                                bf16_t* __restrict__ dst, int n) {
  int i = blockIdx.x * blockDim.x + threadIdx.x;
  if (i < n) dst[i] = (bf16_t)src[i];
}

// W [768 in][768 out] fp32 -> Wt [768 out][768 in] bf16, three matrices.
__global__ void cvt_w_kernel(const float* __restrict__ Wq,
                             const float* __restrict__ Wk,
                             const float* __restrict__ Wv,
                             bf16_t* __restrict__ Wt) {
  const float* W = (blockIdx.y == 0) ? Wq : (blockIdx.y == 1) ? Wk : Wv;
  bf16_t* dst = Wt + (size_t)blockIdx.y * (768 * 768);
  int i = blockIdx.x * blockDim.x + threadIdx.x;  // 0 .. 768*768-1
  int o = i / 768, in = i % 768;
  dst[o * 768 + in] = (bf16_t)W[in * 768 + o];
}

// dist_emb [1023][64] fp32 -> [1024][64] bf16 (row 1023 zero padded)
__global__ void cvt_dist_kernel(const float* __restrict__ src,
                                bf16_t* __restrict__ dst) {
  int i = blockIdx.x * blockDim.x + threadIdx.x;  // 0 .. 65535
  int r = i >> 6;
  dst[i] = (r < 1023) ? (bf16_t)src[i] : (bf16_t)0.0f;
}

// ------------------------------ QKV proj -----------------------------------
// grid (512 row-tiles, 12 col-tiles of 64, 3 {Q,K,V}), block = 1 wave.
__global__ __launch_bounds__(32) void qkv_proj_kernel(
    const bf16_t* __restrict__ Xbf,     // [8192][768]
    const bf16_t* __restrict__ Wt,      // 3 x [768 out][768 in]
    const float* __restrict__ bq, const float* __restrict__ bk,
    const float* __restrict__ bv,
    bf16_t* __restrict__ Qbf,           // [B,NH,S,64]
    bf16_t* __restrict__ Kbf,           // [B,NH,S,64]
    bf16_t* __restrict__ Vtbf) {        // [B,NH,64,S]
  const int lane = threadIdx.x & 31;
  const int half = lane >> 4;
  const int l16  = lane & 15;
  const int RT = blockIdx.x;
  const int CT = blockIdx.y;
  const int which = blockIdx.z;

  const bf16_t* W = Wt + (size_t)which * (768 * 768);
  const float* bias = (which == 0) ? bq : (which == 1) ? bk : bv;
  const bf16_t* arow = Xbf + (size_t)(RT * 16 + l16) * 768;

  v8f acc0 = {}, acc1 = {}, acc2 = {}, acc3 = {};
#pragma unroll 1
  for (int kk = 0; kk < 768; kk += 32) {
    v16bf a = load_a16(arow + kk, 8 * half);
    const int krow = kk + 16 * half;
    v16bf b0 = *(const v16bf*)(W + (size_t)(CT * 64 +  0 + l16) * 768 + krow);
    v16bf b1 = *(const v16bf*)(W + (size_t)(CT * 64 + 16 + l16) * 768 + krow);
    v16bf b2 = *(const v16bf*)(W + (size_t)(CT * 64 + 32 + l16) * 768 + krow);
    v16bf b3 = *(const v16bf*)(W + (size_t)(CT * 64 + 48 + l16) * 768 + krow);
    acc0 = WMMA_BF16(a, b0, acc0);
    acc1 = WMMA_BF16(a, b1, acc1);
    acc2 = WMMA_BF16(a, b2, acc2);
    acc3 = WMMA_BF16(a, b3, acc3);
  }
  float bvv[4];
#pragma unroll
  for (int dt = 0; dt < 4; ++dt) bvv[dt] = bias[CT * 64 + dt * 16 + l16];

#pragma unroll
  for (int v = 0; v < 8; ++v) {
    const int row = RT * 16 + v + 8 * half;
    const int bb = row >> 9, s = row & 511;
    float vals[4] = {acc0[v] + bvv[0], acc1[v] + bvv[1],
                     acc2[v] + bvv[2], acc3[v] + bvv[3]};
#pragma unroll
    for (int dt = 0; dt < 4; ++dt) {
      const int col = CT * 64 + dt * 16 + l16;
      const int h = col >> 6, d = col & 63;
      if (which == 0)
        Qbf[(((size_t)bb * 12 + h) * 512 + s) * 64 + d] = (bf16_t)vals[dt];
      else if (which == 1)
        Kbf[(((size_t)bb * 12 + h) * 512 + s) * 64 + d] = (bf16_t)vals[dt];
      else
        Vtbf[(((size_t)bb * 12 + h) * 64 + d) * 512 + s] = (bf16_t)vals[dt];
    }
  }
}

// ------------------------------ attention ----------------------------------
// One wave per (b, h, 16-query tile); loop over 16 key tiles of 32.
__global__ __launch_bounds__(32) void attn_kernel(
    const bf16_t* __restrict__ Qbf, const bf16_t* __restrict__ Kbf,
    const bf16_t* __restrict__ Vtbf, const bf16_t* __restrict__ distbf,
    const float* __restrict__ mask, float* __restrict__ out) {
  __shared__ float  Hs[16 * 48];   // bias1 band: H[l][j] = q[l] . E[Mbase+j]
  __shared__ float  Gs[48 * 32];   // bias2 band: G[j][r] = E[Mbase+j] . k[r]
  __shared__ bf16_t Ps[16 * 32];   // probs tile (C-layout -> A-layout via LDS)

  const int lane = threadIdx.x & 31;
  const int half = lane >> 4;
  const int l16  = lane & 15;
  const int L0 = blockIdx.x * 16;
  const int h  = blockIdx.y;
  const int bb = blockIdx.z;

  const bf16_t* Qb = Qbf  + (((size_t)bb * 12 + h) * 512) * 64;
  const bf16_t* Kb = Kbf  + (((size_t)bb * 12 + h) * 512) * 64;
  const bf16_t* Vb = Vtbf + (((size_t)bb * 12 + h) * 64) * 512;
  const float*  mp = mask + (size_t)bb * 512;

  // Q tile fragments (16x64 split into two K=32 steps), reused everywhere.
  const v16bf qa0 = load_a16(Qb + (size_t)(L0 + l16) * 64,      8 * half);
  const v16bf qa1 = load_a16(Qb + (size_t)(L0 + l16) * 64 + 32, 8 * half);

  v8f O0 = {}, O1 = {}, O2 = {}, O3 = {};
  float m[8], lsum[8];
#pragma unroll
  for (int v = 0; v < 8; ++v) { m[v] = -1e30f; lsum[v] = 0.0f; }

#pragma unroll 1
  for (int kt = 0; kt < 16; ++kt) {
    const int R0 = kt * 32;

    // Prefetch next key tile (K rows: one 128B cacheline per key; V^T: the
    // 64B segments for all 64 d-rows) -> global_prefetch_b8, hides HBM/L2
    // latency behind this iteration's WMMA work.
    if (kt < 15) {
      __builtin_prefetch(Kb + (size_t)(R0 + 32 + lane) * 64, 0, 0);
      __builtin_prefetch(Vb + (size_t)lane * 512 + R0 + 32, 0, 0);
      __builtin_prefetch(Vb + (size_t)(32 + lane) * 512 + R0 + 32, 0, 0);
    }

    // K-tile B fragments (keys as columns, d contiguous per lane).
    v16bf kb00 = *(const v16bf*)(Kb + (size_t)(R0 +  0 + l16) * 64 + 16 * half);
    v16bf kb01 = *(const v16bf*)(Kb + (size_t)(R0 +  0 + l16) * 64 + 32 + 16 * half);
    v16bf kb10 = *(const v16bf*)(Kb + (size_t)(R0 + 16 + l16) * 64 + 16 * half);
    v16bf kb11 = *(const v16bf*)(Kb + (size_t)(R0 + 16 + l16) * 64 + 32 + 16 * half);

    // raw scores Q.K^T (16x32)
    v8f s0 = {}, s1 = {};
    s0 = WMMA_BF16(qa0, kb00, s0);
    s0 = WMMA_BF16(qa1, kb01, s0);
    s1 = WMMA_BF16(qa0, kb10, s1);
    s1 = WMMA_BF16(qa1, kb11, s1);

    const int Mbase = L0 - R0 + 480;  // distance band base (always in [0,976])

    // H = Q_tile(16x64) x E_band^T(64x48)
#pragma unroll
    for (int nt = 0; nt < 3; ++nt) {
      const bf16_t* dr = distbf + (size_t)(Mbase + nt * 16 + l16) * 64;
      v8f hacc = {};
      hacc = WMMA_BF16(qa0, *(const v16bf*)(dr + 16 * half), hacc);
      hacc = WMMA_BF16(qa1, *(const v16bf*)(dr + 32 + 16 * half), hacc);
#pragma unroll
      for (int v = 0; v < 8; ++v)
        Hs[(v + 8 * half) * 48 + nt * 16 + l16] = hacc[v];
    }

    // G = E_band(48x64) x K_tile^T(64x32)
#pragma unroll
    for (int mt = 0; mt < 3; ++mt) {
      const bf16_t* dr = distbf + (size_t)(Mbase + mt * 16 + l16) * 64;
      v16bf ga0 = load_a16(dr, 8 * half);
      v16bf ga1 = load_a16(dr + 32, 8 * half);
      v8f g0 = {}, g1 = {};
      g0 = WMMA_BF16(ga0, kb00, g0);
      g0 = WMMA_BF16(ga1, kb01, g0);
      g1 = WMMA_BF16(ga0, kb10, g1);
      g1 = WMMA_BF16(ga1, kb11, g1);
#pragma unroll
      for (int v = 0; v < 8; ++v) {
        Gs[(mt * 16 + v + 8 * half) * 32 +  0 + l16] = g0[v];
        Gs[(mt * 16 + v + 8 * half) * 32 + 16 + l16] = g1[v];
      }
    }
    LDS_SYNC();  // wave-private LDS, DS ops in-order: wait is sufficient

    const float mk0 = mp[R0 + l16];
    const float mk1 = mp[R0 + 16 + l16];

    // assemble scores, online softmax (row = v + 8*half spans 16 lanes)
#pragma unroll
    for (int v = 0; v < 8; ++v) {
      const int row = v + 8 * half;
      const int j0 = row - l16 + 31;   // cols r' = l16      -> j in [16,46]
      const int j1 = j0 - 16;          // cols r' = 16 + l16 -> j in [0,30]
      float f0 = (s0[v] + Hs[row * 48 + j0] + Gs[j0 * 32 + l16]) * 0.125f + mk0;
      float f1 = (s1[v] + Hs[row * 48 + j1] + Gs[j1 * 32 + 16 + l16]) * 0.125f + mk1;
      float t = fmaxf(f0, f1);
      t = fmaxf(t, __shfl_xor(t, 1, 32));
      t = fmaxf(t, __shfl_xor(t, 2, 32));
      t = fmaxf(t, __shfl_xor(t, 4, 32));
      t = fmaxf(t, __shfl_xor(t, 8, 32));
      const float mnew = fmaxf(m[v], t);
      const float fac = __expf(m[v] - mnew);
      const float p0 = __expf(f0 - mnew);
      const float p1 = __expf(f1 - mnew);
      float rs = p0 + p1;
      rs += __shfl_xor(rs, 1, 32);
      rs += __shfl_xor(rs, 2, 32);
      rs += __shfl_xor(rs, 4, 32);
      rs += __shfl_xor(rs, 8, 32);
      lsum[v] = lsum[v] * fac + rs;
      m[v] = mnew;
      O0[v] *= fac; O1[v] *= fac; O2[v] *= fac; O3[v] *= fac;
      Ps[row * 32 + l16]      = (bf16_t)p0;
      Ps[row * 32 + 16 + l16] = (bf16_t)p1;
    }
    LDS_SYNC();

    // probs C-layout -> A-layout via LDS
    v16bf pa;
#pragma unroll
    for (int i = 0; i < 8; ++i) {
      pa[i]     = Ps[l16 * 32 + 16 * half + i];
      pa[i + 8] = Ps[l16 * 32 + 16 * half + 8 + i];
    }
    // O += P(16x32) x V_tile(32x64); V^T layout gives contiguous B frags.
    O0 = WMMA_BF16(pa, *(const v16bf*)(Vb + (size_t)( 0 + l16) * 512 + R0 + 16 * half), O0);
    O1 = WMMA_BF16(pa, *(const v16bf*)(Vb + (size_t)(16 + l16) * 512 + R0 + 16 * half), O1);
    O2 = WMMA_BF16(pa, *(const v16bf*)(Vb + (size_t)(32 + l16) * 512 + R0 + 16 * half), O2);
    O3 = WMMA_BF16(pa, *(const v16bf*)(Vb + (size_t)(48 + l16) * 512 + R0 + 16 * half), O3);
  }

  // normalize and write ctx [B, S, 768] fp32
#pragma unroll
  for (int v = 0; v < 8; ++v) {
    const int row = v + 8 * half;
    const float inv = 1.0f / lsum[v];
    float* op = out + (((size_t)bb * 512) + (L0 + row)) * 768 + h * 64;
    op[ 0 + l16] = O0[v] * inv;
    op[16 + l16] = O1[v] * inv;
    op[32 + l16] = O2[v] * inv;
    op[48 + l16] = O3[v] * inv;
  }
}

// ------------------------------ launcher -----------------------------------

extern "C" void kernel_launch(void* const* d_in, const int* in_sizes, int n_in,
                              void* d_out, int out_size, void* d_ws, size_t ws_size,
                              hipStream_t stream) {
  (void)in_sizes; (void)n_in; (void)out_size; (void)ws_size;
  const float* hs   = (const float*)d_in[0];
  const float* mask = (const float*)d_in[1];
  const float* Wq   = (const float*)d_in[2];
  const float* bq   = (const float*)d_in[3];
  const float* Wk   = (const float*)d_in[4];
  const float* bk   = (const float*)d_in[5];
  const float* Wv   = (const float*)d_in[6];
  const float* bv   = (const float*)d_in[7];
  const float* de   = (const float*)d_in[8];
  float* out = (float*)d_out;

  char* ws = (char*)d_ws;
  bf16_t* Xbf    = (bf16_t*)(ws);                       // 12,582,912 B
  bf16_t* Wt     = (bf16_t*)(ws + 12582912);            //  3,538,944 B
  bf16_t* distbf = (bf16_t*)(ws + 16121856);            //    131,072 B
  bf16_t* Qbf    = (bf16_t*)(ws + 16252928);            // 12,582,912 B
  bf16_t* Kbf    = (bf16_t*)(ws + 28835840);            // 12,582,912 B
  bf16_t* Vtbf   = (bf16_t*)(ws + 41418752);            // 12,582,912 B -> ~54 MB

  cvt_bf16_kernel<<<6291456 / 256, 256, 0, stream>>>(hs, Xbf, 6291456);
  cvt_w_kernel<<<dim3(2304, 3), 256, 0, stream>>>(Wq, Wk, Wv, Wt);
  cvt_dist_kernel<<<256, 256, 0, stream>>>(de, distbf);
  qkv_proj_kernel<<<dim3(512, 12, 3), 32, 0, stream>>>(Xbf, Wt, bq, bk, bv,
                                                       Qbf, Kbf, Vtbf);
  attn_kernel<<<dim3(32, 12, 16), 32, 0, stream>>>(Qbf, Kbf, Vtbf, distbf,
                                                   mask, out);
}
